// BaselineAttention_81570018886168
// MI455X (gfx1250) — matrix operand order repaired and checked by reading, not compiled, not run on hardware
//
#include <hip/hip_runtime.h>
#include <hip/hip_bf16.h>
#include <math.h>

// Problem constants (match reference)
#define BB 2
#define NN 2048
#define DM 1024
#define NH 16
#define DH 64
#define KS 32
#define MM (BB * NN)     // 4096 rows for the projection GEMMs
#define KT (DM / 32)     // 32 K-steps of 32 for every GEMM here
#define MG 2             // M-tiles per block (shared A panel in LDS)

typedef __attribute__((ext_vector_type(16))) __bf16       v16bf;
typedef __attribute__((ext_vector_type(8)))  float        v8f;
typedef __attribute__((ext_vector_type(8)))  unsigned int v8u;
typedef __attribute__((ext_vector_type(4)))  unsigned int v4u;
typedef __attribute__((ext_vector_type(8)))  int          v8i;
typedef __attribute__((ext_vector_type(4)))  int          v4i;

__device__ __forceinline__ unsigned short f2bf(float f) {
  unsigned int u = __float_as_uint(f);
  u += 0x7FFFu + ((u >> 16) & 1u); // round-to-nearest-even
  return (unsigned short)(u >> 16);
}

// ---------------------------------------------------------------------------
// Pack fp32 [Mdim, K=1024] matrix into WMMA A-fragment layout (bf16 pairs):
//   Ap[((tileM*KT + kt)*32 + lane)*8 + j]  holds A[m][kk],A[m][kk+1]
//   m = tileM*16 + (lane&15); kbase = (lane>=16)?8:0
//   kk = kt*32 + kbase + (j<4 ? 2j : 16+2(j-4))
// ---------------------------------------------------------------------------
__global__ void k_pack_a(const float* __restrict__ X,
                         unsigned int* __restrict__ Ap) {
  int idx = blockIdx.x * 256 + threadIdx.x;
  int j    = idx & 7;
  int lane = (idx >> 3) & 31;
  int kt   = (idx >> 8) & (KT - 1);
  int tm   = idx >> 13;
  int m  = tm * 16 + (lane & 15);
  int kb = (lane >= 16) ? 8 : 0;
  int kk = kt * 32 + kb + ((j < 4) ? (2 * j) : (16 + 2 * (j - 4)));
  const float* s = X + (size_t)m * DM + kk;
  Ap[idx] = (unsigned)f2bf(s[0]) | ((unsigned)f2bf(s[1]) << 16);
}

// ---------------------------------------------------------------------------
// Pack fp32 weight W[out=1024, in=1024] into WMMA B-fragment layout of W^T:
//   Bp[((tileN*KT + kt)*32 + lane)*8 + j] holds Bt[k][n],Bt[k+1][n] = W[n][k..k+1]
//   n = tileN*16 + (lane&15); kbase = (lane>=16)?8:0; k = kt*32 + kbase + (j<4 ? 2j : 16+2(j-4))
// ---------------------------------------------------------------------------
__global__ void k_pack_b(const float* __restrict__ W,
                         unsigned int* __restrict__ Bp) {
  int idx = blockIdx.x * 256 + threadIdx.x;
  int j    = idx & 7;
  int lane = (idx >> 3) & 31;
  int kt   = (idx >> 8) & (KT - 1);
  int tn   = idx >> 13;
  int n  = tn * 16 + (lane & 15);
  int kb = (lane >= 16) ? 8 : 0;
  int k  = kt * 32 + kb + ((j < 4) ? (2 * j) : (16 + 2 * (j - 4)));
  const float* s = W + (size_t)n * DM + k;  // W row n is contiguous in k
  Bp[idx] = (unsigned)f2bf(s[0]) | ((unsigned)f2bf(s[1]) << 16);
}

__global__ void k_zero(float* __restrict__ p, int n) {
  int i = blockIdx.x * blockDim.x + threadIdx.x;
  if (i < n) p[i] = 0.0f;
}

// ---------------------------------------------------------------------------
// WMMA GEMM on pre-packed fragments.
// Block = 8 waves sharing MG=2 M-tiles (A panel in LDS, 64 KB, one TDM DMA).
// Each wave computes a 2(M) x 4(N) grid of 16x16 tiles -> 8 WMMAs per K-step,
// reusing each B fragment across 2 A fragments and vice versa:
//   per step/wave: 4x ds_load_b128 (A) + 8x global_load_b128 (B) -> 8 WMMA.
// ---------------------------------------------------------------------------
__global__ void k_gemm_packed(const unsigned int* __restrict__ Ap,
                              const unsigned int* __restrict__ Bp,
                              const float* __restrict__ bias,
                              float* __restrict__ C,
                              int Ndim) {
  __shared__ unsigned int sA[MG * KT * 32 * 8]; // 64 KB packed A panels

  const int lane = threadIdx.x & 31;
  const int wave = threadIdx.x >> 5;
  const int nbPerRow = Ndim >> 9;               // blocks per M row (512 cols/blk)
  const int tileM0 = (blockIdx.x / nbPerRow) * MG;
  const int nblk   = blockIdx.x % nbPerRow;
  const int tileN0 = nblk * 32 + wave * 4;      // 4 N-tiles per wave

  // --- TDM: stage MG packed A panels (contiguous) into LDS (wave 0 only) ---
  if (threadIdx.x < 32) {
    const unsigned int* apanel = Ap + (size_t)tileM0 * (KT * 32 * 8);
    unsigned long long ga = (unsigned long long)(const void*)apanel;
    unsigned lds_off = (unsigned)(size_t)&sA[0];
    const unsigned nElems = MG * KT * 32 * 8;   // 16384 x 4-byte elements

    v4u g0;
    g0[0] = 1u;                                  // count=1, user descriptor
    g0[1] = lds_off;                             // lds_addr (bytes)
    g0[2] = (unsigned)(ga & 0xFFFFFFFFu);        // global_addr[31:0]
    g0[3] = (unsigned)((ga >> 32) & 0x01FFFFFFu) // global_addr[56:32]
            | (2u << 30);                        // type = 2 ("image")
    v8i g1;
    g1[0] = (int)(2u << 16);                     // wg_mask=0, data_size=2 (4B)
    g1[1] = (int)((nElems & 0xFFFFu) << 16);     // tensor_dim0[15:0]
    g1[2] = (int)((nElems >> 16) & 0xFFFFu);     // tensor_dim0[31:16], dim1=0
    g1[3] = (int)((nElems & 0xFFFFu) << 16);     // tile_dim0 = nElems (1-D tile)
    g1[4] = 0;                                   // tile_dim1 = tile_dim2 = 0
    g1[5] = (int)nElems;                         // tensor_dim0_stride
    g1[6] = 0;
    g1[7] = 0;
    v4i g2 = {};
    v4i g3 = {};
    v8i g4 = {};
    __builtin_amdgcn_tensor_load_to_lds(g0, g1, g2, g3, g4, 0);
    __builtin_amdgcn_s_wait_tensorcnt(0);
  }
  __syncthreads();

  const size_t bTileStride = (size_t)KT * 32 * 8; // dwords per packed N-tile
  const unsigned int* bb = Bp + (size_t)tileN0 * bTileStride + (size_t)lane * 8;

  v8f acc[MG][4];
#pragma unroll
  for (int mt = 0; mt < MG; ++mt)
#pragma unroll
    for (int t = 0; t < 4; ++t) acc[mt][t] = (v8f){};

#pragma unroll 4
  for (int kt = 0; kt < KT; ++kt) {
    v16bf a[MG];
#pragma unroll
    for (int mt = 0; mt < MG; ++mt) {
      v8u au = *(const v8u*)&sA[(size_t)mt * (KT * 256) + (kt * 32 + lane) * 8];
      a[mt] = __builtin_bit_cast(v16bf, au);
    }
    const unsigned int* bk = bb + (size_t)kt * 256;  // kt*32lanes*8dw
#pragma unroll
    for (int t = 0; t < 4; ++t) {
      v8u bu = *(const v8u*)(bk + (size_t)t * bTileStride);
      v16bf bfr = __builtin_bit_cast(v16bf, bu);
#pragma unroll
      for (int mt = 0; mt < MG; ++mt) {
        acc[mt][t] = __builtin_amdgcn_wmma_f32_16x16x32_bf16(
            false, a[mt], false, bfr, (short)0, acc[mt][t], false, false);
      }
    }
  }

  // C/D layout: vgpr r, lane -> row = r + 8*(lane>>4), col = lane&15
  const int hi = lane >> 4;
#pragma unroll
  for (int mt = 0; mt < MG; ++mt) {
    const int rowBase = (tileM0 + mt) * 16 + 8 * hi;
#pragma unroll
    for (int t = 0; t < 4; ++t) {
      int ncol = (tileN0 + t) * 16 + (lane & 15);
      float bval = bias ? bias[ncol] : 0.0f;
#pragma unroll
      for (int r = 0; r < 8; ++r) {
        C[(size_t)(rowBase + r) * Ndim + ncol] = acc[mt][t][r] + bval;
      }
    }
  }
}

// ---------------- sparse top-k attention row kernel ----------------
#define NTH 256
__global__ void k_attn(const float* __restrict__ Q,
                       const float* __restrict__ Kb,
                       const float* __restrict__ Vb,
                       float* __restrict__ ctx,
                       float* __restrict__ tw,  // [B,H,N,KS] weights
                       int*   __restrict__ ti)  // [B,H,N,KS] indices
{
  __shared__ float qv[DH];
  __shared__ float sc[NN];
  __shared__ float redv[NTH];
  __shared__ int   redi[NTH];
  __shared__ float topv[KS];
  __shared__ int   topi[KS];
  __shared__ float w[KS];

  const int bid = blockIdx.x;            // b*(NH*NN) + h*NN + q
  const int q = bid & (NN - 1);
  const int h = (bid >> 11) & (NH - 1);
  const int b = bid >> 15;
  const int tid = threadIdx.x;

  const float* qp = Q + ((size_t)(b * NN + q) * DM + h * DH);
  if (tid < DH) qv[tid] = qp[tid];
  __syncthreads();

  const float scale = 0.125f; // 1/sqrt(64)
  for (int j = tid; j < NN; j += NTH) {
    float s = -__builtin_inff();
    if (j <= q) { // causal
      const float* kp = Kb + ((size_t)(b * NN + j) * DM + h * DH);
      float a = 0.f;
#pragma unroll 8
      for (int d = 0; d < DH; ++d) a += qv[d] * kp[d];
      s = a * scale;
    }
    sc[j] = s;
  }
  __syncthreads();

  for (int it = 0; it < KS; ++it) {
    float mv = -__builtin_inff();
    int   mi = tid;
    for (int j = tid; j < NN; j += NTH) {
      float v = sc[j];
      if (v > mv) { mv = v; mi = j; }
    }
    redv[tid] = mv; redi[tid] = mi;
    __syncthreads();
    for (int s2 = NTH / 2; s2 > 0; s2 >>= 1) {
      if (tid < s2 && redv[tid + s2] > redv[tid]) {
        redv[tid] = redv[tid + s2];
        redi[tid] = redi[tid + s2];
      }
      __syncthreads();
    }
    if (tid == 0) {
      topv[it] = redv[0];
      topi[it] = redi[0];
      sc[redi[0]] = -__builtin_inff();
    }
    __syncthreads();
  }

  if (tid == 0) {
    float m = topv[0];
    float sum = 0.f;
    for (int i = 0; i < KS; ++i) {
      float e = expf(topv[i] - m);
      w[i] = e;
      sum += e;
    }
    float inv = 1.0f / sum;
    for (int i = 0; i < KS; ++i) w[i] *= inv;
  }
  __syncthreads();

  if (tid < DH) {
    float a = 0.f;
#pragma unroll 4
    for (int i = 0; i < KS; ++i) {
      a += w[i] * Vb[(size_t)(b * NN + topi[i]) * DM + h * DH + tid];
    }
    ctx[(size_t)(b * NN + q) * DM + h * DH + tid] = a;
  }
  if (tid < KS) {
    size_t base = (((size_t)b * NH + h) * NN + q) * KS;
    tw[base + tid] = w[tid];
    ti[base + tid] = topi[tid];
  }
}

// Deterministic attn.mean(heads): one thread owns one (b,q) row of am.
__global__ void k_mean(const float* __restrict__ tw,
                       const int* __restrict__ ti,
                       float* __restrict__ am) {
  int t = blockIdx.x * blockDim.x + threadIdx.x; // b*NN + q
  int b = t >> 11;
  int q = t & (NN - 1);
  float* row = am + ((size_t)b * NN + q) * NN;
  const float invH = 1.0f / NH;
  for (int h = 0; h < NH; ++h) {
    size_t base = (((size_t)b * NH + h) * NN + q) * KS;
    for (int i = 0; i < KS; ++i) {
      float wv = tw[base + i];
      if (wv != 0.f) row[ti[base + i]] += wv * invH;
    }
  }
}

// ---------------- launcher ----------------
extern "C" void kernel_launch(void* const* d_in, const int* in_sizes, int n_in,
                              void* d_out, int out_size, void* d_ws, size_t ws_size,
                              hipStream_t stream) {
  const float* x  = (const float*)d_in[0];
  const float* Wq = (const float*)d_in[1];
  const float* bq = (const float*)d_in[2];
  const float* Wk = (const float*)d_in[3];
  const float* bk = (const float*)d_in[4];
  const float* Wv = (const float*)d_in[5];
  const float* bv = (const float*)d_in[6];
  const float* Wo = (const float*)d_in[7];
  const float* bo = (const float*)d_in[8];

  float* y  = (float*)d_out;                 // [B,N,DM]
  float* am = y + (size_t)BB * NN * DM;      // [B,N,N]

  const size_t nApDw = (size_t)(MM / 16) * KT * 32 * 8;  // 2M dwords (8 MB)
  const size_t nBpDw = (size_t)(DM / 16) * KT * 32 * 8;  // 512K dwords (2 MB)

  char* p = (char*)d_ws;
  unsigned int* xp  = (unsigned int*)p; p += nApDw * 4;
  unsigned int* wqp = (unsigned int*)p; p += nBpDw * 4;
  unsigned int* wkp = (unsigned int*)p; p += nBpDw * 4;
  unsigned int* wvp = (unsigned int*)p; p += nBpDw * 4;
  unsigned int* wop = (unsigned int*)p; p += nBpDw * 4;
  float* Qb   = (float*)p; p += (size_t)MM * DM * 4;
  float* Kbuf = (float*)p; p += (size_t)MM * DM * 4;
  float* Vbuf = (float*)p; p += (size_t)MM * DM * 4;
  float* ctx  = (float*)p; p += (size_t)MM * DM * 4;
  unsigned int* cxp = (unsigned int*)p; p += nApDw * 4;
  float* tw   = (float*)p; p += (size_t)BB * NH * NN * KS * 4;
  int*   ti   = (int*)p;   p += (size_t)BB * NH * NN * KS * 4;

  const int gemmBlocks = (MM / (16 * MG)) * (DM / 512);  // 128 * 2 = 256

  k_pack_a<<<(int)(nApDw / 256), 256, 0, stream>>>(x, xp);
  k_pack_b<<<(int)(nBpDw / 256), 256, 0, stream>>>(Wq, wqp);
  k_pack_b<<<(int)(nBpDw / 256), 256, 0, stream>>>(Wk, wkp);
  k_pack_b<<<(int)(nBpDw / 256), 256, 0, stream>>>(Wv, wvp);
  k_pack_b<<<(int)(nBpDw / 256), 256, 0, stream>>>(Wo, wop);

  k_gemm_packed<<<gemmBlocks, 256, 0, stream>>>(xp, wqp, bq, Qb, DM);
  k_gemm_packed<<<gemmBlocks, 256, 0, stream>>>(xp, wkp, bk, Kbuf, DM);
  k_gemm_packed<<<gemmBlocks, 256, 0, stream>>>(xp, wvp, bv, Vbuf, DM);

  k_zero<<<(BB * NN * NN) / 256, 256, 0, stream>>>(am, BB * NN * NN);

  k_attn<<<BB * NH * NN, NTH, 0, stream>>>(Qb, Kbuf, Vbuf, ctx, tw, ti);
  k_mean<<<(BB * NN) / 256, 256, 0, stream>>>(tw, ti, am);

  k_pack_a<<<(int)(nApDw / 256), 256, 0, stream>>>(ctx, cxp);
  k_gemm_packed<<<gemmBlocks, 256, 0, stream>>>(cxp, wop, bo, y, DM);
}
